// GNNAgent_31147102830761
// MI455X (gfx1250) — matrix operand order repaired and checked by reading, not compiled, not run on hardware
//
#include <hip/hip_runtime.h>
#include <stdint.h>

// ---------------- problem constants ----------------
constexpr int N_AGENTS  = 64;
constexpr int HIDDEN    = 256;
constexpr int IN_DIM    = 128;
constexpr int N_ACTIONS = 16;
constexpr int B_EP      = 4096;
constexpr int ROWS      = B_EP * N_AGENTS; // 262144

// ---------------- WMMA types ----------------
typedef __attribute__((ext_vector_type(16))) __bf16 v16bf;
typedef __attribute__((ext_vector_type(8)))  float  v8f;

struct FragAB { union { v16bf v; uint4 q[2]; }; };

__device__ __forceinline__ unsigned short f2bf(float f) {
    uint32_t u = __builtin_bit_cast(uint32_t, f);
    u += 0x7FFFu + ((u >> 16) & 1u);          // round-to-nearest-even
    return (unsigned short)(u >> 16);
}

__device__ __forceinline__ v8f wmma_bf16(v16bf a, v16bf b, v8f c) {
    return __builtin_amdgcn_wmma_f32_16x16x32_bf16(false, a, false, b, (short)0, c, false, false);
}

// A fragment: row-major bf16 buffer, contiguous k order, same as the B loaders (lane m=lane&15, K = half*16 .. +16 contiguous)
__device__ __forceinline__ v16bf load_a_lds(const unsigned short* buf, int stride,
                                            int mBase, int kTile, int lane) {
    int m = lane & 15, half = lane >> 4;
    const unsigned short* p = buf + (mBase + m) * stride + kTile + half * 16;
    FragAB f;
    f.q[0] = *reinterpret_cast<const uint4*>(p);
    f.q[1] = *reinterpret_cast<const uint4*>(p + 8);
    return f.v;
}

// B fragment from transposed bf16 buffer (row = N index): lane n=lane&15, K = half*16 .. +16 contiguous
__device__ __forceinline__ v16bf load_b_lds(const unsigned short* buf, int stride,
                                            int nBase, int kTile, int lane) {
    int n = lane & 15, half = lane >> 4;
    const unsigned short* p = buf + (nBase + n) * stride + kTile + half * 16;
    FragAB f;
    f.q[0] = *reinterpret_cast<const uint4*>(p);
    f.q[1] = *reinterpret_cast<const uint4*>(p + 8);
    return f.v;
}

__device__ __forceinline__ v16bf load_b_gl(const unsigned short* __restrict__ buf, int strideK,
                                           int nBase, int kTile, int lane) {
    int n = lane & 15, half = lane >> 4;
    const unsigned short* p = buf + (size_t)(nBase + n) * strideK + kTile + half * 16;
    FragAB f;
    f.q[0] = *reinterpret_cast<const uint4*>(p);
    f.q[1] = *reinterpret_cast<const uint4*>(p + 8);
    return f.v;
}

// ---------------- LDS layout (ushort units) ----------------
constexpr int STR_COMM = 72;   // 64 + 8 pad
constexpr int STR_X    = 136;  // 128 + 8
constexpr int STR_RM   = 264;  // 256 + 8
constexpr int STR_T    = 72;   // 64 + 8

constexpr int L_COMM  = 0;
constexpr int L_X     = L_COMM  + 64  * STR_COMM;  // 4608
constexpr int L_ENC   = L_X     + 64  * STR_X;     // 13312
constexpr int L_MSGT  = L_ENC   + 64  * STR_RM;    // 30208
constexpr int L_AGG   = L_MSGT  + 256 * STR_T;     // 48640
constexpr int L_MSGRM = L_AGG   + 64  * STR_RM;    // 65536
constexpr int L_TOTAL = L_MSGRM + 64  * STR_RM;    // 82432 ushorts
constexpr int SMEM_BYTES = L_TOTAL * 2;            // 164864 B (< 320KB WGP LDS, 1 WG/WGP)

// ---------------- prep: comm = D_hat @ A_hat @ D_hat (bf16) ----------------
__global__ void k_comm(const float* __restrict__ adj, unsigned short* __restrict__ comm_bf) {
    __shared__ float dinv[N_AGENTS];
    int i = threadIdx.x;                      // 64 threads
    float s = 1.0f;                           // self loop
    for (int j = 0; j < N_AGENTS; ++j) s += adj[i * N_AGENTS + j];
    dinv[i] = rsqrtf(s);
    __syncthreads();
    float di = dinv[i];
    for (int j = 0; j < N_AGENTS; ++j) {
        float a = adj[i * N_AGENTS + j] + ((i == j) ? 1.0f : 0.0f);
        comm_bf[i * N_AGENTS + j] = f2bf(di * a * dinv[j]);
    }
}

// ---------------- prep: transpose+convert weights fp32[K][N] -> bf16[N][K] ----------------
__global__ void k_transpose_bf16(const float* __restrict__ src, unsigned short* __restrict__ dst,
                                 int K, int N) {
    int idx = blockIdx.x * blockDim.x + threadIdx.x;
    if (idx >= K * N) return;
    int k = idx / N, n = idx - k * N;
    dst[n * K + k] = f2bf(src[idx]);
}

// ---------------- fused per-episode GNN forward ----------------
__global__ void __launch_bounds__(256, 1)
gnn_fused(const float* __restrict__ x,
          const unsigned short* __restrict__ comm_bf,
          const unsigned short* __restrict__ enc_wt,   // [256][128]
          const unsigned short* __restrict__ msg_wt,   // [256][256]
          const unsigned short* __restrict__ pol_wt,   // [256][512]
          const unsigned short* __restrict__ act_wt,   // [16][256]
          const float* __restrict__ enc_b,
          const float* __restrict__ pol_b,
          const float* __restrict__ act_b,
          float* __restrict__ out_actions,             // [ROWS][16]
          float* __restrict__ out_h)                   // [ROWS][256]
{
    extern __shared__ unsigned short smem[];
    unsigned short* sComm  = smem + L_COMM;
    unsigned short* sX     = smem + L_X;
    unsigned short* sEnc   = smem + L_ENC;
    unsigned short* sMsgT  = smem + L_MSGT;
    unsigned short* sAgg   = smem + L_AGG;   // reused as hRM after pol stage
    unsigned short* sMsgRM = smem + L_MSGRM;

    const int tid  = threadIdx.x;
    const int lane = tid & 31;
    const int wave = tid >> 5;
    const int n    = lane & 15;
    const int half = lane >> 4;
    const int ep   = blockIdx.x;
    const int r0   = ep * N_AGENTS;

    const int mBase = (wave & 3) * 16;
    const int cg    = wave >> 2;             // 0 or 1 (column half of 256)

    if (tid == 0) {  // warm L2 with weight lines (global_prefetch)
        __builtin_prefetch(enc_wt, 0, 1);
        __builtin_prefetch(msg_wt, 0, 1);
        __builtin_prefetch(pol_wt, 0, 1);
        __builtin_prefetch(act_wt, 0, 1);
    }

    // ---- stage in comm + x (fp32 -> bf16) ----
    for (int idx = tid; idx < N_AGENTS * N_AGENTS; idx += 256) {
        int i = idx >> 6, j = idx & 63;
        sComm[i * STR_COMM + j] = comm_bf[idx];
    }
    for (int idx = tid; idx < N_AGENTS * IN_DIM; idx += 256) {
        int i = idx >> 7, j = idx & 127;
        sX[i * STR_X + j] = f2bf(x[(size_t)(r0 + i) * IN_DIM + j]);
    }
    __syncthreads();

    // ---- encoder: enc = relu(x @ enc_w + b)  (M64 N256 K128) ----
    for (int g = 0; g < 2; ++g) {            // 2 groups of 4 column tiles (reg pressure)
        v8f acc[4];
        #pragma unroll
        for (int t = 0; t < 4; ++t) acc[t] = (v8f){};
        for (int ks = 0; ks < 4; ++ks) {
            int kT = ks * 32;
            v16bf a = load_a_lds(sX, STR_X, mBase, kT, lane);
            #pragma unroll
            for (int t = 0; t < 4; ++t) {
                v16bf b = load_b_gl(enc_wt, IN_DIM, cg * 128 + g * 64 + t * 16, kT, lane);
                acc[t] = wmma_bf16(a, b, acc[t]);
            }
        }
        #pragma unroll
        for (int t = 0; t < 4; ++t) {
            int nB = cg * 128 + g * 64 + t * 16;
            float bias = enc_b[nB + n];
            #pragma unroll
            for (int r = 0; r < 8; ++r) {
                int row = mBase + r + half * 8;
                float v = fmaxf(acc[t][r] + bias, 0.0f);
                unsigned short bv = f2bf(v);
                sEnc[row * STR_RM + nB + n]   = bv;       // row-major (pol A)
                sMsgT[(nB + n) * STR_T + row] = bv;       // transposed (layer-1 B)
            }
        }
    }
    __syncthreads();

    // ---- 3 GNN layers ----
    for (int l = 0; l < 3; ++l) {
        // agg = comm @ msg  (M64 N256 K64): A=comm rowmajor LDS, B=msgT LDS
        for (int g = 0; g < 2; ++g) {
            v8f acc[4];
            #pragma unroll
            for (int t = 0; t < 4; ++t) acc[t] = (v8f){};
            #pragma unroll
            for (int ks = 0; ks < 2; ++ks) {
                int kT = ks * 32;
                v16bf a = load_a_lds(sComm, STR_COMM, mBase, kT, lane);
                #pragma unroll
                for (int t = 0; t < 4; ++t) {
                    v16bf b = load_b_lds(sMsgT, STR_T, cg * 128 + g * 64 + t * 16, kT, lane);
                    acc[t] = wmma_bf16(a, b, acc[t]);
                }
            }
            #pragma unroll
            for (int t = 0; t < 4; ++t) {
                int nB = cg * 128 + g * 64 + t * 16;
                #pragma unroll
                for (int r = 0; r < 8; ++r) {
                    int row = mBase + r + half * 8;
                    sAgg[row * STR_RM + nB + n] = f2bf(acc[t][r]);
                }
            }
        }
        __syncthreads();

        // msg = relu(agg @ msg_w)  (M64 N256 K256): A=agg LDS, B=msg_wt global
        for (int g = 0; g < 2; ++g) {
            v8f acc[4];
            #pragma unroll
            for (int t = 0; t < 4; ++t) acc[t] = (v8f){};
            for (int ks = 0; ks < 8; ++ks) {
                int kT = ks * 32;
                v16bf a = load_a_lds(sAgg, STR_RM, mBase, kT, lane);
                #pragma unroll
                for (int t = 0; t < 4; ++t) {
                    v16bf b = load_b_gl(msg_wt, HIDDEN, cg * 128 + g * 64 + t * 16, kT, lane);
                    acc[t] = wmma_bf16(a, b, acc[t]);
                }
            }
            #pragma unroll
            for (int t = 0; t < 4; ++t) {
                int nB = cg * 128 + g * 64 + t * 16;
                #pragma unroll
                for (int r = 0; r < 8; ++r) {
                    int row = mBase + r + half * 8;
                    unsigned short bv = f2bf(fmaxf(acc[t][r], 0.0f));
                    if (l < 2) sMsgT[(nB + n) * STR_T + row] = bv;   // B for next layer
                    else       sMsgRM[row * STR_RM + nB + n] = bv;   // A for pol concat
                }
            }
        }
        __syncthreads();
    }

    // ---- pol: h = relu([enc|msg] @ pol_w + b)  (M64 N256 K512) ----
    for (int g = 0; g < 2; ++g) {
        v8f acc[4];
        #pragma unroll
        for (int t = 0; t < 4; ++t) acc[t] = (v8f){};
        for (int ks = 0; ks < 16; ++ks) {
            int kT = ks * 32;
            v16bf a = (kT < 256) ? load_a_lds(sEnc,   STR_RM, mBase, kT,       lane)
                                 : load_a_lds(sMsgRM, STR_RM, mBase, kT - 256, lane);
            #pragma unroll
            for (int t = 0; t < 4; ++t) {
                v16bf b = load_b_gl(pol_wt, 2 * HIDDEN, cg * 128 + g * 64 + t * 16, kT, lane);
                acc[t] = wmma_bf16(a, b, acc[t]);
            }
        }
        #pragma unroll
        for (int t = 0; t < 4; ++t) {
            int nB = cg * 128 + g * 64 + t * 16;
            float bias = pol_b[nB + n];
            #pragma unroll
            for (int r = 0; r < 8; ++r) {
                int row = mBase + r + half * 8;
                float v = fmaxf(acc[t][r] + bias, 0.0f);
                out_h[(size_t)(r0 + row) * HIDDEN + nB + n] = v;   // fp32 output
                sAgg[row * STR_RM + nB + n] = f2bf(v);             // hRM (act A), reuse sAgg
            }
        }
    }
    __syncthreads();

    // ---- act: actions = h @ act_w + b  (M64 N16 K256), waves 0-3 ----
    if (wave < 4) {
        int mB = wave * 16;
        v8f acc = (v8f){};
        for (int ks = 0; ks < 8; ++ks) {
            int kT = ks * 32;
            v16bf a = load_a_lds(sAgg, STR_RM, mB, kT, lane);
            v16bf b = load_b_gl(act_wt, HIDDEN, 0, kT, lane);
            acc = wmma_bf16(a, b, acc);
        }
        float bias = act_b[n];
        #pragma unroll
        for (int r = 0; r < 8; ++r) {
            int row = mB + r + half * 8;
            out_actions[(size_t)(r0 + row) * N_ACTIONS + n] = acc[r] + bias;
        }
    }
}

// ---------------- host launcher ----------------
extern "C" void kernel_launch(void* const* d_in, const int* in_sizes, int n_in,
                              void* d_out, int out_size, void* d_ws, size_t ws_size,
                              hipStream_t stream) {
    const float* x     = (const float*)d_in[0];
    const float* adj   = (const float*)d_in[1];
    const float* enc_w = (const float*)d_in[2];
    const float* enc_b = (const float*)d_in[3];
    const float* msg_w = (const float*)d_in[4];
    const float* pol_w = (const float*)d_in[5];
    const float* pol_b = (const float*)d_in[6];
    const float* act_w = (const float*)d_in[7];
    const float* act_b = (const float*)d_in[8];

    // workspace layout (bytes)
    char* ws = (char*)d_ws;
    unsigned short* comm_bf = (unsigned short*)(ws + 0);        //  8192 B
    unsigned short* enc_wt  = (unsigned short*)(ws + 8192);     // 65536 B  [256][128]
    unsigned short* msg_wt  = (unsigned short*)(ws + 73728);    // 131072 B [256][256]
    unsigned short* pol_wt  = (unsigned short*)(ws + 204800);   // 262144 B [256][512]
    unsigned short* act_wt  = (unsigned short*)(ws + 466944);   //  8192 B  [16][256]

    float* out_actions = (float*)d_out;                         // [ROWS][16]
    float* out_h       = out_actions + (size_t)ROWS * N_ACTIONS;// [ROWS][256]

    k_comm<<<1, 64, 0, stream>>>(adj, comm_bf);
    k_transpose_bf16<<<(IN_DIM * HIDDEN + 255) / 256, 256, 0, stream>>>(enc_w, enc_wt, IN_DIM, HIDDEN);
    k_transpose_bf16<<<(HIDDEN * HIDDEN + 255) / 256, 256, 0, stream>>>(msg_w, msg_wt, HIDDEN, HIDDEN);
    k_transpose_bf16<<<(2 * HIDDEN * HIDDEN + 255) / 256, 256, 0, stream>>>(pol_w, pol_wt, 2 * HIDDEN, HIDDEN);
    k_transpose_bf16<<<(HIDDEN * N_ACTIONS + 255) / 256, 256, 0, stream>>>(act_w, act_wt, HIDDEN, N_ACTIONS);

    (void)hipFuncSetAttribute((const void*)gnn_fused,
                              hipFuncAttributeMaxDynamicSharedMemorySize, SMEM_BYTES);
    gnn_fused<<<B_EP, 256, SMEM_BYTES, stream>>>(x, comm_bf, enc_wt, msg_wt, pol_wt, act_wt,
                                                 enc_b, pol_b, act_b, out_actions, out_h);
}
